// StructureModule_52183852646503
// MI455X (gfx1250) — matrix operand order repaired and checked by reading, not compiled, not run on hardware
//
#include <hip/hip_runtime.h>
#include <cstdint>

// ============================================================================
// AlphaFold-style StructureModule forward for MI455X (gfx1250, wave32).
//
// Roofline: z (512x512x128 fp32 = 128MB) dominates traffic. Strategy:
//  * LayerNorm(z) -> bf16 once (64MB) -> resident in the 192MB L2 for the
//    8 per-iteration opair GEMMs; bias = LN(z)@Wb hoisted out of the loop.
//  * Every matmul (~28 GFLOP total) goes through v_wmma_f32_16x16x32_bf16
//    (fp32 accumulate). The point-attention cross term is folded into the
//    q.k logits GEMM by concatenating coef-scaled global points (K=160).
//  * GEMM fragments are unconditional 16B vector loads (out-of-range rows/
//    cols only pollute discarded outputs; ws is tail-padded so reads stay
//    in mapped memory). Guards exist only on stores.
// ============================================================================

typedef __bf16 bf16;
typedef __attribute__((ext_vector_type(16))) __bf16 v16bf;
typedef __attribute__((ext_vector_type(8)))  __bf16 v8bf;
typedef __attribute__((ext_vector_type(8)))  float  v8f;

#define DEV __device__ __forceinline__
#define SEQ     512
#define HEADS   12
#define QPTS    4
#define VPTS    8
#define W_Cc    0.2357022603955158f   // sqrt(2/(9*QP))
#define W_Lc    0.5773502691896258f   // sqrt(1/3)
#define RSCALE  0.08838834764831845f  // 128^-0.5

DEV float wred(float v) {
#pragma unroll
  for (int m = 16; m > 0; m >>= 1) v += __shfl_xor(v, m, 32);
  return v;
}

DEV v16bf cat16(v8bf lo, v8bf hi) {
  return __builtin_shufflevector(lo, hi, 0, 1, 2, 3, 4, 5, 6, 7,
                                 8, 9, 10, 11, 12, 13, 14, 15);
}

// ---------------------------------------------------------------------------
// Batched WMMA GEMM: C[b] = op(A[b]) @ B[b] (+bias, relu, accumulate).
// A row-major MxK. B either "BT" (NxK row-major) or row-major KxN.
// One 16x16 tile per wave, 8 waves/block, K multiple of 32 at all call sites.
// Fragment layouts used here (wave32; A/B use contiguous k, same order in both; C/D per CDNA5 ISA 7.12.2):
//   A lanes<16: K 0..15        ; lanes>=16: K 16..31
//   B lanes<16: K 0..15        ; lanes>=16: K 16..31
//   C/D lanes<16: rows 0..7    ; lanes>=16: rows 8..15 (col = lane&15)
// ---------------------------------------------------------------------------
struct GemmArgs {
  const void* A; const void* B; float* C; const float* bias;
  long lda, ldb, ldc, aBatch, bBatch, cBatch;
  int M, N, K;
  int reluA, reluOut, accumC;
};

template <int ABF, int BBF, int BTR>
__global__ void wmma_gemm(GemmArgs g) {
  const int wave = threadIdx.x >> 5, lane = threadIdx.x & 31;
  const int tilesN = (g.N + 15) >> 4;
  const int tilesM = (g.M + 15) >> 4;
  const int T = blockIdx.x * 8 + wave;
  if (T >= tilesM * tilesN) return;          // wave-uniform: EXEC all-1 for WMMA
  const int tN = (T % tilesN) << 4;
  const int tM = (T / tilesN) << 4;
  const long aOff = (long)blockIdx.z * g.aBatch;
  const long bOff = (long)blockIdx.z * g.bBatch;
  const long cOff = (long)blockIdx.z * g.cBatch;

  const int half = lane >> 4, lidx = lane & 15;
  const int arow = tM + lidx;
  const int bn   = tN + lidx;
  const bool bval = bn < g.N;
  const int reluA = g.reluA;

  v8f acc = {};
  for (int kk = 0; kk < g.K; kk += 32) {
    const int k0  = kk + half * 16;
    const int k1  = kk + half * 16 + 8;
    const int bk0 = kk + half * 16;
    v16bf af, bfv;

    if constexpr (ABF) {
      const bf16* A = (const bf16*)g.A + aOff + (long)arow * g.lda;
      af = cat16(*(const v8bf*)(A + k0), *(const v8bf*)(A + k1));
    } else {
      const float* A = (const float*)g.A + aOff + (long)arow * g.lda;
      float4 x0 = *(const float4*)(A + k0);
      float4 x1 = *(const float4*)(A + k0 + 4);
      float4 y0 = *(const float4*)(A + k1);
      float4 y1 = *(const float4*)(A + k1 + 4);
      if (reluA) {
        x0.x = fmaxf(x0.x, 0.f); x0.y = fmaxf(x0.y, 0.f); x0.z = fmaxf(x0.z, 0.f); x0.w = fmaxf(x0.w, 0.f);
        x1.x = fmaxf(x1.x, 0.f); x1.y = fmaxf(x1.y, 0.f); x1.z = fmaxf(x1.z, 0.f); x1.w = fmaxf(x1.w, 0.f);
        y0.x = fmaxf(y0.x, 0.f); y0.y = fmaxf(y0.y, 0.f); y0.z = fmaxf(y0.z, 0.f); y0.w = fmaxf(y0.w, 0.f);
        y1.x = fmaxf(y1.x, 0.f); y1.y = fmaxf(y1.y, 0.f); y1.z = fmaxf(y1.z, 0.f); y1.w = fmaxf(y1.w, 0.f);
      }
      af[0]  = (bf16)x0.x; af[1]  = (bf16)x0.y; af[2]  = (bf16)x0.z; af[3]  = (bf16)x0.w;
      af[4]  = (bf16)x1.x; af[5]  = (bf16)x1.y; af[6]  = (bf16)x1.z; af[7]  = (bf16)x1.w;
      af[8]  = (bf16)y0.x; af[9]  = (bf16)y0.y; af[10] = (bf16)y0.z; af[11] = (bf16)y0.w;
      af[12] = (bf16)y1.x; af[13] = (bf16)y1.y; af[14] = (bf16)y1.z; af[15] = (bf16)y1.w;
    }

    if constexpr (BTR) {
      if constexpr (BBF) {
        const bf16* B = (const bf16*)g.B + bOff + (long)bn * g.ldb;
        bfv = cat16(*(const v8bf*)(B + bk0), *(const v8bf*)(B + bk0 + 8));
      } else {
        const float* B = (const float*)g.B + bOff + (long)bn * g.ldb;
        float4 x0 = *(const float4*)(B + bk0);
        float4 x1 = *(const float4*)(B + bk0 + 4);
        float4 y0 = *(const float4*)(B + bk0 + 8);
        float4 y1 = *(const float4*)(B + bk0 + 12);
        bfv[0]  = (bf16)x0.x; bfv[1]  = (bf16)x0.y; bfv[2]  = (bf16)x0.z; bfv[3]  = (bf16)x0.w;
        bfv[4]  = (bf16)x1.x; bfv[5]  = (bf16)x1.y; bfv[6]  = (bf16)x1.z; bfv[7]  = (bf16)x1.w;
        bfv[8]  = (bf16)y0.x; bfv[9]  = (bf16)y0.y; bfv[10] = (bf16)y0.z; bfv[11] = (bf16)y0.w;
        bfv[12] = (bf16)y1.x; bfv[13] = (bf16)y1.y; bfv[14] = (bf16)y1.z; bfv[15] = (bf16)y1.w;
      }
    } else {  // row-major KxN: column gather (coalesced across the 16 lanes)
      if constexpr (BBF) {
        const bf16* B = (const bf16*)g.B + bOff + bn;
#pragma unroll
        for (int e = 0; e < 16; e++) bfv[e] = B[(long)(bk0 + e) * g.ldb];
      } else {
        const float* B = (const float*)g.B + bOff + bn;
#pragma unroll
        for (int e = 0; e < 16; e++) bfv[e] = (bf16)B[(long)(bk0 + e) * g.ldb];
      }
    }

    acc = __builtin_amdgcn_wmma_f32_16x16x32_bf16(false, af, false, bfv,
                                                  (short)0, acc, false, false);
  }

  const float bcol = (g.bias && bval) ? g.bias[bn] : 0.f;
  const int mbase = half * 8;
  float* C = g.C + cOff;
#pragma unroll
  for (int r = 0; r < 8; r++) {
    const int row = tM + mbase + r;
    if (row < g.M && bval) {
      const long off = (long)row * g.ldc + bn;
      float v = acc[r] + bcol;
      if (g.accumC)  v += C[off];
      if (g.reluOut) v = fmaxf(v, 0.f);
      C[off] = v;
    }
  }
}

// --------------------------- LayerNorm kernels -----------------------------
__global__ void ln_rows(const float* x, float* y, const float* g, const float* b, int nrows) {
  const int row = blockIdx.x * 8 + (threadIdx.x >> 5);
  const int lane = threadIdx.x & 31;
  if (row >= nrows) return;
  const float4 v = reinterpret_cast<const float4*>(x + (long)row * 128)[lane];
  float s = wred(v.x + v.y + v.z + v.w);
  const float mean = s * (1.f / 128.f);
  const float dx = v.x - mean, dy = v.y - mean, dz = v.z - mean, dw = v.w - mean;
  float q = wred(dx*dx + dy*dy + dz*dz + dw*dw);
  const float rstd = rsqrtf(q * (1.f / 128.f) + 1e-5f);
  const float4 gv = reinterpret_cast<const float4*>(g)[lane];
  const float4 bv = reinterpret_cast<const float4*>(b)[lane];
  float4 o; o.x = dx*rstd*gv.x + bv.x; o.y = dy*rstd*gv.y + bv.y;
  o.z = dz*rstd*gv.z + bv.z; o.w = dw*rstd*gv.w + bv.w;
  reinterpret_cast<float4*>(y + (long)row * 128)[lane] = o;
}

__global__ void zln_bf16(const float* z, bf16* out, const float* g, const float* b) {
  const long row = (long)blockIdx.x * 8 + (threadIdx.x >> 5);
  const int lane = threadIdx.x & 31;
  const float4 v = reinterpret_cast<const float4*>(z + row * 128)[lane];
  float s = wred(v.x + v.y + v.z + v.w);
  const float mean = s * (1.f / 128.f);
  const float dx = v.x - mean, dy = v.y - mean, dz = v.z - mean, dw = v.w - mean;
  float q = wred(dx*dx + dy*dy + dz*dz + dw*dw);
  const float rstd = rsqrtf(q * (1.f / 128.f) + 1e-5f);
  const float4 gv = reinterpret_cast<const float4*>(g)[lane];
  const float4 bv = reinterpret_cast<const float4*>(b)[lane];
  v8bf* o = (v8bf*)(out + row * 128 + lane * 4);  // write 4 bf16 = 8B
  bf16 tmp[4] = {(bf16)(dx*rstd*gv.x + bv.x), (bf16)(dy*rstd*gv.y + bv.y),
                 (bf16)(dz*rstd*gv.z + bv.z), (bf16)(dw*rstd*gv.w + bv.w)};
  bf16* ob = out + row * 128 + lane * 4;
  ob[0] = tmp[0]; ob[1] = tmp[1]; ob[2] = tmp[2]; ob[3] = tmp[3];
  (void)o;
}

__global__ void add_ln(float* s, const float* h, const float* g, const float* b) {
  const int row = blockIdx.x * 8 + (threadIdx.x >> 5);
  const int lane = threadIdx.x & 31;
  float4 v = reinterpret_cast<const float4*>(s + (long)row * 128)[lane];
  const float4 hv = reinterpret_cast<const float4*>(h + (long)row * 128)[lane];
  v.x += hv.x; v.y += hv.y; v.z += hv.z; v.w += hv.w;
  float sum = wred(v.x + v.y + v.z + v.w);
  const float mean = sum * (1.f / 128.f);
  const float dx = v.x - mean, dy = v.y - mean, dz = v.z - mean, dw = v.w - mean;
  float q = wred(dx*dx + dy*dy + dz*dz + dw*dw);
  const float rstd = rsqrtf(q * (1.f / 128.f) + 1e-5f);
  const float4 gv = reinterpret_cast<const float4*>(g)[lane];
  const float4 bv = reinterpret_cast<const float4*>(b)[lane];
  float4 o; o.x = dx*rstd*gv.x + bv.x; o.y = dy*rstd*gv.y + bv.y;
  o.z = dz*rstd*gv.z + bv.z; o.w = dw*rstd*gv.w + bv.w;
  reinterpret_cast<float4*>(s + (long)row * 128)[lane] = o;
}

__global__ void add_inplace(float* x, const float* y, long n) {
  const long i = (long)blockIdx.x * 256 + threadIdx.x;
  if (i < n) x[i] += y[i];
}

__global__ void w_tr_bf16(const float* W, bf16* WT, int K, int N) {
  const long idx = (long)blockIdx.x * 256 + threadIdx.x;
  if (idx >= (long)K * N) return;
  const long n = idx / K, k = idx % K;
  WT[idx] = (bf16)W[k * (long)N + n];
}

// ----------------------- pack q/k (+points) for logits ----------------------
__global__ void pack_qk(const float* qv, const float* pl, const float* R, const float* t,
                        const float* gamma, bf16* cat, float* ss, float rs, int applyCoef) {
  const int i = blockIdx.x, tid = threadIdx.x;
  for (int idx = tid; idx < 1536; idx += 256) {
    const int h = idx >> 7;
    cat[((long)h * SEQ + i) * 160 + (idx & 127)] = (bf16)(qv[(long)i * 1536 + idx] * rs);
  }
  if (tid < HEADS) {
    const int h = tid;
    const float coef = applyCoef ? log1pf(expf(gamma[h])) * W_Cc : 1.f;
    const float* Ri = R + i * 9; const float* ti = t + i * 3;
    bf16* dst = cat + ((long)h * SEQ + i) * 160 + 128;
    float acc = 0.f;
#pragma unroll
    for (int p = 0; p < QPTS; p++) {
      const float* pp = pl + (long)i * (HEADS * QPTS * 3) + (h * QPTS + p) * 3;
      const float x = pp[0], y = pp[1], z = pp[2];
      const float gx = Ri[0]*x + Ri[1]*y + Ri[2]*z + ti[0];
      const float gy = Ri[3]*x + Ri[4]*y + Ri[5]*z + ti[1];
      const float gz = Ri[6]*x + Ri[7]*y + Ri[8]*z + ti[2];
      acc += gx*gx + gy*gy + gz*gz;
      dst[p*3+0] = (bf16)(gx*coef); dst[p*3+1] = (bf16)(gy*coef); dst[p*3+2] = (bf16)(gz*coef);
    }
#pragma unroll
    for (int e = 12; e < 32; e++) dst[e] = (bf16)0.f;   // zero K-pad to 160
    ss[i * HEADS + h] = acc;
  }
}

__global__ void pack_vp(const float* pl, const float* R, const float* t, float* vg) {
  const int i = blockIdx.x, tid = threadIdx.x;
  if (tid >= HEADS * VPTS) return;
  const int h = tid >> 3, p = tid & 7;
  const float* Ri = R + i * 9; const float* ti = t + i * 3;
  const float* pp = pl + (long)i * (HEADS * VPTS * 3) + (h * VPTS + p) * 3;
  const float x = pp[0], y = pp[1], z = pp[2];
  float* o = vg + (long)i * (HEADS * VPTS * 3) + (h * VPTS + p) * 3;
  o[0] = Ri[0]*x + Ri[1]*y + Ri[2]*z + ti[0];
  o[1] = Ri[3]*x + Ri[4]*y + Ri[5]*z + ti[1];
  o[2] = Ri[6]*x + Ri[7]*y + Ri[8]*z + ti[2];
}

// ------------------- score epilogue + softmax over i ------------------------
__global__ void score_softmax(float* a, const float* bias16, const float* qq,
                              const float* kks, const float* gamma) {
  const int h = blockIdx.y;
  const int jl = threadIdx.x, il = threadIdx.y;
  const int j = blockIdx.x * 16 + jl;
  const float coef = log1pf(expf(gamma[h])) * W_Cc;
  const float kkj = kks[j * HEADS + h];
  float vals[16]; float mx = -1e30f;
#pragma unroll
  for (int r = 0; r < 16; r++) {
    const int i = il + r * 32;
    float v = a[((long)h * SEQ + i) * SEQ + j];
    v += bias16[((long)i * SEQ + j) * 16 + h];
    v -= 0.5f * coef * (qq[i * HEADS + h] + kkj);
    vals[r] = v * W_Lc; mx = fmaxf(mx, vals[r]);
  }
  __shared__ float red[32][16];
  red[il][jl] = mx; __syncthreads();
  for (int st = 16; st > 0; st >>= 1) {
    if (il < st) red[il][jl] = fmaxf(red[il][jl], red[il + st][jl]);
    __syncthreads();
  }
  mx = red[0][jl]; __syncthreads();
  float sum = 0.f;
#pragma unroll
  for (int r = 0; r < 16; r++) { vals[r] = expf(vals[r] - mx); sum += vals[r]; }
  red[il][jl] = sum; __syncthreads();
  for (int st = 16; st > 0; st >>= 1) {
    if (il < st) red[il][jl] += red[il + st][jl];
    __syncthreads();
  }
  const float inv = 1.f / red[0][jl];
#pragma unroll
  for (int r = 0; r < 16; r++)
    a[((long)h * SEQ + (il + r * 32)) * SEQ + j] = vals[r] * inv;
}

__global__ void rowsum(const float* a, float* arow) {
  const int row = blockIdx.x * 8 + (threadIdx.x >> 5);
  const int lane = threadIdx.x & 31;
  const float* r = a + (long)row * SEQ;
  float s = 0.f;
#pragma unroll
  for (int e = 0; e < 16; e++) s += r[lane + e * 32];
  s = wred(s);
  if (lane == 0) arow[row] = s;
}

__global__ void o_into_cat(const float* arow, const float* v, float* cat) {
  const long idx = (long)blockIdx.x * 256 + threadIdx.x;
  if (idx >= (long)SEQ * 1536) return;
  const long i = idx / 1536, hc = idx % 1536, h = hc >> 7;
  cat[i * 3456 + hc] = arow[h * SEQ + i] * v[idx];
}

__global__ void onorm_cat(float* cat) {
  const int idx = blockIdx.x * 256 + threadIdx.x;
  if (idx >= SEQ * HEADS * VPTS) return;
  const int i = idx / 96, hp = idx % 96, h = hp / 8, p = hp % 8;
  const float* b = cat + (long)i * 3456 + 3072 + h * 24 + p * 3;
  cat[(long)i * 3456 + 3360 + hp] = fmaxf(sqrtf(b[0]*b[0] + b[1]*b[1] + b[2]*b[2]), 1e-6f);
}

// -------------------------- backbone update ---------------------------------
__global__ void bb_update_k(const float* s,
                            const float* Wb, const float* bb, const float* Wc, const float* bc,
                            const float* Wd, const float* bd, const float* Wt, const float* bt,
                            float* R, float* t) {
  const int i = blockIdx.x, lane = threadIdx.x;
  const float* sr = s + (long)i * 128;
  float pb = 0, pc = 0, pd = 0, p0 = 0, p1 = 0, p2 = 0;
#pragma unroll
  for (int e = 0; e < 4; e++) {
    const int k = lane * 4 + e; const float x = sr[k];
    pb += x * Wb[k]; pc += x * Wc[k]; pd += x * Wd[k];
    p0 += x * Wt[k*3]; p1 += x * Wt[k*3+1]; p2 += x * Wt[k*3+2];
  }
  pb = wred(pb); pc = wred(pc); pd = wred(pd);
  p0 = wred(p0); p1 = wred(p1); p2 = wred(p2);
  if (lane == 0) {
    const float qb = pb + bb[0], qc = pc + bc[0], qd = pd + bd[0];
    const float tu0 = p0 + bt[0], tu1 = p1 + bt[1], tu2 = p2 + bt[2];
    const float inv = rsqrtf(1.f + qb*qb + qc*qc + qd*qd);
    const float w = inv, x = qb*inv, y = qc*inv, z = qd*inv;
    float Ru[9] = {1 - 2*(y*y + z*z), 2*(x*y - w*z),     2*(x*z + w*y),
                   2*(x*y + w*z),     1 - 2*(x*x + z*z), 2*(y*z - w*x),
                   2*(x*z - w*y),     2*(y*z + w*x),     1 - 2*(x*x + y*y)};
    float* Rp = R + i * 9; float* tp = t + i * 3;
    float Ro[9], to[3];
#pragma unroll
    for (int r = 0; r < 3; r++) {
#pragma unroll
      for (int c = 0; c < 3; c++)
        Ro[r*3+c] = Rp[r*3]*Ru[c] + Rp[r*3+1]*Ru[3+c] + Rp[r*3+2]*Ru[6+c];
      to[r] = tp[r] + Rp[r*3]*tu0 + Rp[r*3+1]*tu1 + Rp[r*3+2]*tu2;
    }
#pragma unroll
    for (int e = 0; e < 9; e++) Rp[e] = Ro[e];
    tp[0] = to[0]; tp[1] = to[1]; tp[2] = to[2];
  }
}

__global__ void init_rt(float* R, float* t) {
  const int i = blockIdx.x * 256 + threadIdx.x;
  if (i >= SEQ) return;
  float* r = R + i * 9;
  r[0]=1.f; r[1]=0.f; r[2]=0.f; r[3]=0.f; r[4]=1.f; r[5]=0.f; r[6]=0.f; r[7]=0.f; r[8]=1.f;
  t[i*3]=0.f; t[i*3+1]=0.f; t[i*3+2]=0.f;
}

__global__ void anorm_k(const float* una, float* an) {
  const int idx = blockIdx.x * 256 + threadIdx.x;
  if (idx >= SEQ * 7) return;
  const int i = idx / 7, p = idx % 7;
  const float a0 = una[i*14 + p*2], a1 = una[i*14 + p*2 + 1];
  const float inv = rsqrtf(a0*a0 + a1*a1);
  an[i*14 + p*2] = a0 * inv; an[i*14 + p*2 + 1] = a1 * inv;
}

// Reference's torsion frames compose against a zero rigid (zR==0), so every
// frame translation equals t and pred[i,k,:] = 8*t[i].
__global__ void finalize_k(const float* s, const float* una, const float* an,
                           const float* t, float* out) {
  const int idx = blockIdx.x * 256 + threadIdx.x;
  if (idx >= 102912) return;
  float v;
  if      (idx < 65536) v = s[idx];
  else if (idx < 72704) v = una[idx - 65536];
  else if (idx < 79872) v = an[idx - 72704];
  else if (idx < 81408) v = t[idx - 79872];
  else { const int r = idx - 81408; v = 8.f * t[(r / 42) * 3 + (r % 3)]; }
  out[idx] = v;
}

// ============================ host orchestration ============================
static void gemm(hipStream_t st, const void* A, int aBf, long lda, long aB,
                 const void* B, int bBf, int bTr, long ldb, long bB,
                 float* Cm, long ldc, long cB, int M, int N, int K,
                 const float* bias, int reluA, int reluOut, int accum, int batch) {
  GemmArgs g; g.A = A; g.B = B; g.C = Cm; g.bias = bias;
  g.lda = lda; g.ldb = ldb; g.ldc = ldc; g.aBatch = aB; g.bBatch = bB; g.cBatch = cB;
  g.M = M; g.N = N; g.K = K; g.reluA = reluA; g.reluOut = reluOut; g.accumC = accum;
  const int tiles = ((M + 15) / 16) * ((N + 15) / 16);
  dim3 grid((unsigned)((tiles + 7) / 8), 1, (unsigned)batch);
  if (aBf) {
    // only combo used with bf16 A is bf16 transposed B
    wmma_gemm<1, 1, 1><<<grid, dim3(256), 0, st>>>(g);
  } else if (bTr) {
    wmma_gemm<0, 1, 1><<<grid, dim3(256), 0, st>>>(g);
  } else if (bBf) {
    wmma_gemm<0, 1, 0><<<grid, dim3(256), 0, st>>>(g);
  } else {
    wmma_gemm<0, 0, 0><<<grid, dim3(256), 0, st>>>(g);
  }
}

extern "C" void kernel_launch(void* const* d_in, const int* in_sizes, int n_in,
                              void* d_out, int out_size, void* d_ws, size_t ws_size,
                              hipStream_t stream) {
  (void)in_sizes; (void)n_in; (void)out_size; (void)ws_size;
  const float* s_raw = (const float*)d_in[0];
  const float* z_raw = (const float*)d_in[1];
  auto P = [&](int k) { return (const float*)d_in[3 + k]; };
  // Param leaves: JAX pytree flatten of the params dict (sorted keys; tuples
  // and lists flatten in order):
  //  0 Wb | 1 Wk | 2 Wkp | 3 Wout.W | 4 Wout.b | 5 Wq | 6 Wqp | 7 Wv | 8 Wvp
  //  9..40 a_blocks[b] = (W1,b1,W2,b2)
  //  41 a_in.W 42 a_in.b | 43 a_init.W 44 a_init.b | 45 a_out.W 46 a_out.b
  //  47 bb_b.W 48 bb_b.b | 49 bb_c.W 50 bb_c.b | 51 bb_d.W 52 bb_d.b
  //  53 bb_t.W 54 bb_t.b | 55 gamma | 56 lin_in.W 57 lin_in.b
  //  58 ln_ipa.g 59 ln_ipa.b | 60 ln_s.g 61 ln_s.b | 62 ln_t.g 63 ln_t.b
  //  64 ln_z.g 65 ln_z.b | 66 t1.W 67 t1.b | 68 t2.W 69 t2.b | 70 t3.W 71 t3.b
  const float *Wb = P(0), *Wk = P(1), *Wkp = P(2), *WoutW = P(3), *Woutb = P(4),
              *Wq = P(5), *Wqp = P(6), *Wv = P(7), *Wvp = P(8);
  const float *ainW = P(41), *ainb = P(42), *aiW = P(43), *aib = P(44),
              *aoW = P(45), *aob = P(46);
  const float *bbW = P(47), *bbB = P(48), *bcW = P(49), *bcB = P(50),
              *bdW = P(51), *bdB = P(52), *btW = P(53), *btB = P(54);
  const float *gamma = P(55), *linW = P(56), *linb = P(57);
  const float *lnipa_g = P(58), *lnipa_b = P(59), *lns_g = P(60), *lns_b = P(61),
              *lnt_g = P(62), *lnt_b = P(63), *lnz_g = P(64), *lnz_b = P(65);
  const float *t1W = P(66), *t1b = P(67), *t2W = P(68), *t2b = P(69),
              *t3W = P(70), *t3b = P(71);

  char* wsp = (char*)d_ws;
  auto alloc = [&](size_t bytes) { void* p = wsp; wsp += (bytes + 255) & ~(size_t)255; return p; };
  const long IJ = (long)SEQ * SEQ;
  bf16*  zbf    = (bf16*) alloc(IJ * 128 * 2);
  float* bias16 = (float*)alloc(IJ * 16 * 4);
  float* abuf   = (float*)alloc((long)HEADS * IJ * 4);
  bf16*  qcat   = (bf16*) alloc((long)HEADS * SEQ * 160 * 2);
  bf16*  kcat   = (bf16*) alloc((long)HEADS * SEQ * 160 * 2);
  float* s_cur  = (float*)alloc(SEQ * 128 * 4);
  float* s_init = (float*)alloc(SEQ * 128 * 4);
  float* cat    = (float*)alloc((long)SEQ * 3456 * 4);
  float* qb     = (float*)alloc((long)SEQ * 1536 * 4);
  float* kb     = (float*)alloc((long)SEQ * 1536 * 4);
  float* vb     = (float*)alloc((long)SEQ * 1536 * 4);
  float* qpl    = (float*)alloc(SEQ * 144 * 4);
  float* kpl    = (float*)alloc(SEQ * 144 * 4);
  float* vpl    = (float*)alloc(SEQ * 288 * 4);
  float* vpg    = (float*)alloc(SEQ * 288 * 4);
  float* qq     = (float*)alloc(SEQ * HEADS * 4);
  float* kks    = (float*)alloc(SEQ * HEADS * 4);
  float* arow   = (float*)alloc((long)HEADS * SEQ * 4);
  float* ipa_o  = (float*)alloc(SEQ * 128 * 4);
  float* h1     = (float*)alloc(SEQ * 128 * 4);
  float* h2     = (float*)alloc(SEQ * 128 * 4);
  float* ang    = (float*)alloc(SEQ * 128 * 4);
  float* atmp   = (float*)alloc(SEQ * 128 * 4);
  float* una    = (float*)alloc(SEQ * 14 * 4);
  float* anbuf  = (float*)alloc(SEQ * 14 * 4);
  float* Rbuf   = (float*)alloc(SEQ * 9 * 4);
  float* tbuf   = (float*)alloc(SEQ * 3 * 4);
  auto allocW = [&](long n) { return (bf16*)alloc(n * 2); };
  bf16 *linT = allocW(128L*128), *WqT = allocW(1536L*128), *WkT = allocW(1536L*128),
       *WvT = allocW(1536L*128), *WqpT = allocW(144L*128), *WkpT = allocW(144L*128),
       *WvpT = allocW(288L*128), *WbT = allocW(12L*128), *WoutT = allocW(3456L*128),
       *t1T = allocW(128L*128), *t2T = allocW(128L*128), *t3T = allocW(128L*128),
       *ainT = allocW(128L*128), *aiT = allocW(128L*128), *aoT = allocW(14L*128);
  bf16* blkT[16];
  for (int bi = 0; bi < 16; bi++) blkT[bi] = allocW(128L*128);
  (void)alloc(1 << 16);  // tail pad: guard-free GEMM fragment reads stay mapped

  auto tr = [&](const float* Wm, bf16* WT, int K, int N) {
    const long n = (long)K * N;
    w_tr_bf16<<<dim3((unsigned)((n + 255) / 256)), dim3(256), 0, stream>>>(Wm, WT, K, N);
  };
  tr(linW, linT, 128, 128);
  tr(Wq,  WqT,  128, 1536);
  tr(Wk,  WkT,  128, 1536);
  tr(Wv,  WvT,  128, 1536);
  tr(Wqp, WqpT, 128, 144);
  tr(Wkp, WkpT, 128, 144);
  tr(Wvp, WvpT, 128, 288);
  tr(Wb,  WbT,  128, 12);
  tr(WoutW, WoutT, 3456, 128);
  tr(t1W, t1T, 128, 128);
  tr(t2W, t2T, 128, 128);
  tr(t3W, t3T, 128, 128);
  tr(ainW, ainT, 128, 128);
  tr(aiW,  aiT,  128, 128);
  tr(aoW,  aoT,  128, 14);
  for (int bi = 0; bi < 8; bi++) {
    tr(P(9 + 4*bi + 0), blkT[2*bi + 0], 128, 128);
    tr(P(9 + 4*bi + 2), blkT[2*bi + 1], 128, 128);
  }

  // ---- forward ----
  ln_rows<<<64, 256, 0, stream>>>(s_raw, s_init, lns_g, lns_b, SEQ);
  gemm(stream, s_init, 0, 128, 0, linT, 1, 1, 128, 0, s_cur, 128, 0,
       SEQ, 128, 128, linb, 0, 0, 0, 1);
  zln_bf16<<<32768, 256, 0, stream>>>(z_raw, zbf, lnz_g, lnz_b);
  gemm(stream, zbf, 1, 128, 0, WbT, 1, 1, 128, 0, bias16, 16, 0,
       (int)IJ, 12, 128, nullptr, 0, 0, 0, 1);           // bias hoisted out of loop
  init_rt<<<2, 256, 0, stream>>>(Rbuf, tbuf);

  for (int it = 0; it < 8; it++) {
    gemm(stream, s_cur, 0, 128, 0, WqT,  1, 1, 128, 0, qb,  1536, 0, SEQ, 1536, 128, nullptr, 0,0,0,1);
    gemm(stream, s_cur, 0, 128, 0, WkT,  1, 1, 128, 0, kb,  1536, 0, SEQ, 1536, 128, nullptr, 0,0,0,1);
    gemm(stream, s_cur, 0, 128, 0, WvT,  1, 1, 128, 0, vb,  1536, 0, SEQ, 1536, 128, nullptr, 0,0,0,1);
    gemm(stream, s_cur, 0, 128, 0, WqpT, 1, 1, 128, 0, qpl, 144,  0, SEQ, 144, 128, nullptr, 0,0,0,1);
    gemm(stream, s_cur, 0, 128, 0, WkpT, 1, 1, 128, 0, kpl, 144,  0, SEQ, 144, 128, nullptr, 0,0,0,1);
    gemm(stream, s_cur, 0, 128, 0, WvpT, 1, 1, 128, 0, vpl, 288,  0, SEQ, 288, 128, nullptr, 0,0,0,1);
    pack_qk<<<SEQ, 256, 0, stream>>>(qb, qpl, Rbuf, tbuf, gamma, qcat, qq, RSCALE, 1);
    pack_qk<<<SEQ, 256, 0, stream>>>(kb, kpl, Rbuf, tbuf, gamma, kcat, kks, 1.f, 0);
    pack_vp<<<SEQ, 128, 0, stream>>>(vpl, Rbuf, tbuf, vpg);
    // logits (q.k * c^-1/2 + coef*qp.kp), batched over heads, K=160
    gemm(stream, qcat, 1, 160, (long)SEQ*160, kcat, 1, 1, 160, (long)SEQ*160,
         abuf, SEQ, IJ, SEQ, SEQ, 160, nullptr, 0, 0, 0, HEADS);
    score_softmax<<<dim3(32, HEADS), dim3(16, 32), 0, stream>>>(abuf, bias16, qq, kks, gamma);
    rowsum<<<768, 256, 0, stream>>>(abuf, arow);
    o_into_cat<<<3072, 256, 0, stream>>>(arow, vb, cat);
    // opts into cat[.,3072:3360], batched over heads
    gemm(stream, abuf, 0, SEQ, IJ, vpg, 0, 0, 288, 24,
         cat + 3072, 3456, 24, SEQ, 24, SEQ, nullptr, 0, 0, 0, HEADS);
    onorm_cat<<<192, 256, 0, stream>>>(cat);
    // opair into cat[.,1536:3072], batched over residues (a @ LN(z), L2-resident)
    gemm(stream, abuf, 0, IJ, SEQ, zbf, 1, 0, 128, (long)SEQ*128,
         cat + 1536, 128, 3456, HEADS, 128, SEQ, nullptr, 0, 0, 0, SEQ);
    gemm(stream, cat, 0, 3456, 0, WoutT, 1, 1, 3456, 0, ipa_o, 128, 0,
         SEQ, 128, 3456, Woutb, 0, 0, 0, 1);
    add_inplace<<<256, 256, 0, stream>>>(s_cur, ipa_o, (long)SEQ * 128);
    ln_rows<<<64, 256, 0, stream>>>(s_cur, s_cur, lnipa_g, lnipa_b, SEQ);
    gemm(stream, s_cur, 0, 128, 0, t1T, 1, 1, 128, 0, h1, 128, 0, SEQ, 128, 128, t1b, 0, 1, 0, 1);
    gemm(stream, h1,    0, 128, 0, t2T, 1, 1, 128, 0, h2, 128, 0, SEQ, 128, 128, t2b, 0, 1, 0, 1);
    gemm(stream, h2,    0, 128, 0, t3T, 1, 1, 128, 0, h1, 128, 0, SEQ, 128, 128, t3b, 0, 0, 0, 1);
    add_ln<<<64, 256, 0, stream>>>(s_cur, h1, lnt_g, lnt_b);
    bb_update_k<<<SEQ, 32, 0, stream>>>(s_cur, bbW, bbB, bcW, bcB, bdW, bdB, btW, btB, Rbuf, tbuf);
    // angle resnet
    gemm(stream, s_cur,  0, 128, 0, ainT, 1, 1, 128, 0, ang, 128, 0, SEQ, 128, 128, ainb, 1, 0, 0, 1);
    gemm(stream, s_init, 0, 128, 0, aiT,  1, 1, 128, 0, ang, 128, 0, SEQ, 128, 128, aib,  1, 0, 1, 1);
    for (int bi = 0; bi < 8; bi++) {
      gemm(stream, ang,  0, 128, 0, blkT[2*bi],   1, 1, 128, 0, atmp, 128, 0,
           SEQ, 128, 128, P(9 + 4*bi + 1), 1, 1, 0, 1);
      gemm(stream, atmp, 0, 128, 0, blkT[2*bi+1], 1, 1, 128, 0, ang,  128, 0,
           SEQ, 128, 128, P(9 + 4*bi + 3), 0, 0, 1, 1);
    }
    gemm(stream, ang, 0, 128, 0, aoT, 1, 1, 128, 0, una, 14, 0, SEQ, 14, 128, aob, 1, 0, 0, 1);
    anorm_k<<<14, 256, 0, stream>>>(una, anbuf);
  }
  finalize_k<<<402, 256, 0, stream>>>(s_cur, una, anbuf, tbuf, (float*)d_out);
}
